// SimpleHGNLayers_38439957299975
// MI455X (gfx1250) — matrix operand order repaired and checked by reading, not compiled, not run on hardware
//
#include <hip/hip_runtime.h>
#include <hip/hip_bf16.h>
#include <math.h>

// ---------------- constants from the reference ----------------
#define NN      20000
#define EE      320000
#define IND     256
#define HIDD    64
#define CLSD    16
#define NETD    5
#define EDD     64
#define NTYP    3
#define ALPHA_C 0.05f
#define NEG_C   0.2f

typedef __attribute__((ext_vector_type(16))) _Float16 v16h;
typedef __attribute__((ext_vector_type(8)))  _Float16 v8h;
typedef __attribute__((ext_vector_type(4)))  _Float16 v4h;
typedef __attribute__((ext_vector_type(8)))  float    v8f;

// ---------------- small helpers ----------------
__device__ __forceinline__ void atomicMaxF(float* addr, float val) {
    // IEEE-754 ordered-compare trick: signed-int max for >=0, unsigned-int min for <0
    if (val >= 0.0f) atomicMax((int*)addr, __float_as_int(val));
    else             atomicMin((unsigned int*)addr, __float_as_uint(val));
}

// vectorized fill: n4 = count of float4 elements (b128 stores)
__global__ void k_fill4(float4* __restrict__ p, float v, long long n4) {
    long long i = blockIdx.x * (long long)blockDim.x + threadIdx.x;
    if (i < n4) p[i] = make_float4(v, v, v, v);
}

// vectorized f32 -> f16: n4 = count of 4-element groups
__global__ void k_cvt16(const float* __restrict__ a, _Float16* __restrict__ b, long long n4) {
    long long i = blockIdx.x * (long long)blockDim.x + threadIdx.x;
    if (i >= n4) return;
    float4 f = *(const float4*)(a + i * 4);
    v4h h;
    h[0] = (_Float16)f.x; h[1] = (_Float16)f.y;
    h[2] = (_Float16)f.z; h[3] = (_Float16)f.w;
    *(v4h*)(b + i * 4) = h;
}

// Pack B[K x Nc] (f32, row-major) into per-lane WMMA layout:
// Bp[((nt*ktiles + kt)*32 + lane)*16 + j] = B[kt*32 + (lane>>4)*8 + (j&7) + (j>>3)*16][nt*16 + (lane&15)]
__global__ void k_packB(const float* __restrict__ B, _Float16* __restrict__ Bp, int K, int Nc) {
    int idx   = blockIdx.x * blockDim.x + threadIdx.x;
    int total = (Nc >> 4) * (K >> 5) * 512;
    if (idx >= total) return;
    int j    = idx & 15;
    int lane = (idx >> 4) & 31;
    int kt   = (idx >> 9) % (K >> 5);
    int nt   = (idx >> 9) / (K >> 5);
    int col  = nt * 16 + (lane & 15);
    int k    = kt * 32 + (lane >> 4) * 8 + (j & 7) + (j >> 3) * 16;
    Bp[idx] = (_Float16)B[(size_t)k * Nc + col];
}

// ---------------- WMMA GEMM: C[M x Nc] = A(f16)[M x K] * Bpacked + bias ----------------
// one wave per 16-row tile; grid.y = Nc/16; fused bias / row-type mask / f16 mirror.
__global__ void k_gemm_wmma(const _Float16* __restrict__ A, const _Float16* __restrict__ Bp,
                            float* __restrict__ C, _Float16* __restrict__ C16,
                            const float* __restrict__ bias,
                            const int* __restrict__ rowmask, int maskval,
                            int M, int K, int Nc) {
    const int lane  = threadIdx.x & 31;
    const int wave  = threadIdx.x >> 5;
    const int mtile = blockIdx.x * (blockDim.x >> 5) + wave;
    if (mtile * 16 >= M) return;               // whole wave exits together: EXEC stays full
    const int ntile = blockIdx.y;
    const int m0 = mtile * 16, n0 = ntile * 16;
    const int row   = m0 + (lane & 15);
    const int khalf = lane >> 4;
    const int ktiles = K >> 5;

    const _Float16* ap = A + (size_t)row * K + khalf * 8;
    const _Float16* bp = Bp + (size_t)ntile * ktiles * 512 + lane * 16;

    v8f acc = {};
    for (int kt = 0; kt < ktiles; ++kt) {
        v8h a0 = *(const v8h*)(ap);            // K: kbase .. kbase+7
        v8h a1 = *(const v8h*)(ap + 16);       // K: kbase+16 .. kbase+23
        v16h a = __builtin_shufflevector(a0, a1, 0,1,2,3,4,5,6,7,8,9,10,11,12,13,14,15);
        v16h b = *(const v16h*)(bp);
        acc = __builtin_amdgcn_wmma_f32_16x16x32_f16(false, a, false, b,
                                                     (short)0, acc, false, false);
        ap += 32;
        bp += 512;
    }

    const int col   = n0 + (lane & 15);
    const int rbase = m0 + (lane >> 4) * 8;
    const float bv  = bias ? bias[col] : 0.0f;
#pragma unroll
    for (int r = 0; r < 8; ++r) {
        int mr = rbase + r;
        if (rowmask && rowmask[mr] != maskval) continue;
        float v = acc[r] + bv;
        C[(size_t)mr * Nc + col] = v;
        if (C16) C16[(size_t)mr * Nc + col] = (_Float16)v;
    }
}

// ---------------- attention-side kernels ----------------
// el/er: per (node, head) dot of feat with al/ar
__global__ void k_elr(const float* __restrict__ feat, const float* __restrict__ al,
                      const float* __restrict__ ar, float* __restrict__ el,
                      float* __restrict__ er, int n, int H, int D) {
    int i = blockIdx.x * blockDim.x + threadIdx.x;
    if (i >= n * H) return;
    int node = i / H, h = i % H;
    const float* f = feat + (size_t)node * H * D + h * D;
    const float* a = al + h * D;
    const float* b = ar + h * D;
    float sl = 0.f, sr = 0.f;
    for (int d = 0; d < D; ++d) { float v = f[d]; sl += v * a[d]; sr += v * b[d]; }
    el[i] = sl; er[i] = sr;
}

// ee table: [NET x H] = dot over d2 of (Ee[t] @ We)[h*ED+d2] * ae[h][d2]
__global__ void k_eetab(const float* __restrict__ Ee, const float* __restrict__ We,
                        const float* __restrict__ ae, float* __restrict__ out,
                        int H, int ED_) {
    int t = blockIdx.x, h = threadIdx.x;
    float acc = 0.f;
    for (int d2 = 0; d2 < ED_; ++d2) {
        float f = 0.f;
        for (int d = 0; d < ED_; ++d) f += Ee[t * ED_ + d] * We[(size_t)d * (H * ED_) + h * ED_ + d2];
        acc += f * ae[h * ED_ + d2];
    }
    out[t * H + h] = acc;
}

__global__ void k_logit(const float* __restrict__ el, const float* __restrict__ er,
                        const float* __restrict__ eetab, const int* __restrict__ src,
                        const int* __restrict__ dst, const int* __restrict__ et,
                        float* __restrict__ logit, float* __restrict__ m, int E_, int H) {
    int i = blockIdx.x * blockDim.x + threadIdx.x;
    if (i >= E_ * H) return;
    int e = i / H, h = i % H;
    float L = el[src[e] * H + h] + er[dst[e] * H + h] + eetab[et[e] * H + h];
    L = (L > 0.f) ? L : NEG_C * L;
    logit[i] = L;
    atomicMaxF(&m[dst[e] * H + h], L);
}

__global__ void k_expsum(float* __restrict__ logit, const float* __restrict__ m,
                         const int* __restrict__ dst, float* __restrict__ s, int E_, int H) {
    int i = blockIdx.x * blockDim.x + threadIdx.x;
    if (i >= E_ * H) return;
    int e = i / H, h = i % H;
    float ex = __expf(logit[i] - m[dst[e] * H + h]);
    logit[i] = ex;
    atomicAdd(&s[dst[e] * H + h], ex);
}

// 4-wide over d: one b128 gather of feat[src], indices/attn amortized over 4 outputs.
// Dq = D/4. Optional res-attn blend + attn store (layer-0 only).
__global__ void k_aggregate4(const float* __restrict__ ex, const float* __restrict__ s,
                             const int* __restrict__ src, const int* __restrict__ dst,
                             const float* __restrict__ feat, float* __restrict__ agg,
                             const float* __restrict__ res_attn, float* __restrict__ attn_out,
                             int E_, int H, int Dq) {
    long long i = blockIdx.x * (long long)blockDim.x + threadIdx.x;
    long long tot = (long long)E_ * H * Dq;
    if (i >= tot) return;
    int q  = (int)(i % Dq);
    int eh = (int)(i / Dq);
    int e = eh / H, h = eh % H;
    int dn = dst[e];
    float a = ex[eh] / s[dn * H + h];
    if (res_attn) a = a * (1.f - ALPHA_C) + res_attn[eh] * ALPHA_C;
    if (attn_out && q == 0) attn_out[eh] = a;
    const int D = Dq * 4;
    float4 f = *(const float4*)(feat + (size_t)src[e] * H * D + h * D + q * 4);
    float* o = agg + (size_t)dn * H * D + h * D + q * 4;
    atomicAdd(o + 0, a * f.x);
    atomicAdd(o + 1, a * f.y);
    atomicAdd(o + 2, a * f.z);
    atomicAdd(o + 3, a * f.w);
}

// out = agg + resid + b, optional ELU; mirrors to f16 for next GEMM
__global__ void k_finalize(const float* __restrict__ agg, const float* __restrict__ resid,
                           const float* __restrict__ b, float* __restrict__ hout,
                           _Float16* __restrict__ hout16, int n, int HD, int act) {
    int i = blockIdx.x * blockDim.x + threadIdx.x;
    if (i >= n * HD) return;
    int j = i % HD;
    float v = agg[i] + (resid ? resid[i] : 0.f) + b[j];
    if (act) v = (v > 0.f) ? v : (__expf(v) - 1.f);
    hout[i] = v;
    if (hout16) hout16[i] = (_Float16)v;
}

__global__ void k_norm(const float* __restrict__ lg, float* __restrict__ out, int n, int C) {
    int i = blockIdx.x * blockDim.x + threadIdx.x;
    if (i >= n) return;
    float ss = 0.f;
    for (int c = 0; c < C; ++c) { float v = lg[i * C + c]; ss += v * v; }
    float nr = fmaxf(sqrtf(ss), 1e-12f);
    for (int c = 0; c < C; ++c) out[i * C + c] = lg[i * C + c] / nr;
}

// ---------------- host side ----------------
static inline unsigned gblk(long long n, int b) { return (unsigned)((n + b - 1) / b); }

extern "C" void kernel_launch(void* const* d_in, const int* in_sizes, int n_in,
                              void* d_out, int out_size, void* d_ws, size_t ws_size,
                              hipStream_t stream) {
    (void)in_sizes; (void)n_in; (void)out_size; (void)ws_size;
    const float* x       = (const float*)d_in[0];
    const int* edge_idx  = (const int*)d_in[1];
    const int* etype     = (const int*)d_in[2];
    const int* ntype     = (const int*)d_in[3];
    const float* fc_w    = (const float*)d_in[4];
    const float* fc_b    = (const float*)d_in[5];
    const float* l0_W  = (const float*)d_in[6];  const float* l0_Ee = (const float*)d_in[7];
    const float* l0_We = (const float*)d_in[8];  const float* l0_al = (const float*)d_in[9];
    const float* l0_ar = (const float*)d_in[10]; const float* l0_ae = (const float*)d_in[11];
    const float* l0_b  = (const float*)d_in[12];
    const float* l1_W  = (const float*)d_in[13]; const float* l1_Ee = (const float*)d_in[14];
    const float* l1_We = (const float*)d_in[15]; const float* l1_al = (const float*)d_in[16];
    const float* l1_ar = (const float*)d_in[17]; const float* l1_ae = (const float*)d_in[18];
    const float* l1_b  = (const float*)d_in[19];
    const float* l2_W  = (const float*)d_in[20]; const float* l2_Ee = (const float*)d_in[21];
    const float* l2_We = (const float*)d_in[22]; const float* l2_al = (const float*)d_in[23];
    const float* l2_ar = (const float*)d_in[24]; const float* l2_ae = (const float*)d_in[25];
    const float* l2_b  = (const float*)d_in[26]; const float* l2_rW = (const float*)d_in[27];

    const int* src = edge_idx;
    const int* dst = edge_idx + EE;

    // carve workspace (256B aligned chunks)
    char* ws = (char*)d_ws;
    size_t off = 0;
    auto carve = [&](size_t bytes) -> char* {
        char* p = ws + off;
        off = (off + bytes + 255) & ~(size_t)255;
        return p;
    };
    _Float16* xh   = (_Float16*)carve((size_t)NN * IND * 2);
    float*    h    = (float*)   carve((size_t)NN * 256 * 4);
    _Float16* h16  = (_Float16*)carve((size_t)NN * 256 * 2);
    float*    feat = (float*)   carve((size_t)NN * 256 * 4);
    float*    agg  = (float*)   carve((size_t)NN * 256 * 4);
    float*    res2 = (float*)   carve((size_t)NN * CLSD * 4);
    float*    attn0= (float*)   carve((size_t)EE * 4 * 4);
    float*    elog = (float*)   carve((size_t)EE * 4 * 4);
    float*    elb  = (float*)   carve((size_t)NN * 4 * 4);
    float*    erb  = (float*)   carve((size_t)NN * 4 * 4);
    float*    mbuf = (float*)   carve((size_t)NN * 4 * 4);
    float*    sbuf = (float*)   carve((size_t)NN * 4 * 4);
    float*    eet  = (float*)   carve((size_t)NETD * 4 * 4);
    _Float16* pfc0 = (_Float16*)carve((size_t)IND * HIDD * 2);
    _Float16* pfc1 = (_Float16*)carve((size_t)IND * HIDD * 2);
    _Float16* pfc2 = (_Float16*)carve((size_t)IND * HIDD * 2);
    _Float16* p0   = (_Float16*)carve((size_t)HIDD * 256 * 2);
    _Float16* p1   = (_Float16*)carve((size_t)256 * 256 * 2);
    _Float16* p2   = (_Float16*)carve((size_t)256 * CLSD * 2);
    _Float16* pr   = (_Float16*)carve((size_t)256 * CLSD * 2);

    const int mtiles = NN / 16;               // 1250
    const dim3 gemmBlk(128);                  // 4 waves / block
    const unsigned gx = (mtiles + 3) / 4;     // 313

    // ---- stage 0: conversions + weight packing ----
    k_cvt16<<<gblk((long long)NN * IND / 4, 256), 256, 0, stream>>>(x, xh, (long long)NN * IND / 4);
    k_packB<<<gblk((HIDD/16)*(IND/32)*512, 256), 256, 0, stream>>>(fc_w + 0*IND*HIDD, pfc0, IND, HIDD);
    k_packB<<<gblk((HIDD/16)*(IND/32)*512, 256), 256, 0, stream>>>(fc_w + 1*IND*HIDD, pfc1, IND, HIDD);
    k_packB<<<gblk((HIDD/16)*(IND/32)*512, 256), 256, 0, stream>>>(fc_w + 2*IND*HIDD, pfc2, IND, HIDD);
    k_packB<<<gblk((256/16)*(HIDD/32)*512, 256), 256, 0, stream>>>(l0_W,  p0, HIDD, 256);
    k_packB<<<gblk((256/16)*(256/32)*512, 256), 256, 0, stream>>>(l1_W,  p1, 256, 256);
    k_packB<<<gblk((CLSD/16)*(256/32)*512, 256), 256, 0, stream>>>(l2_W,  p2, 256, CLSD);
    k_packB<<<gblk((CLSD/16)*(256/32)*512, 256), 256, 0, stream>>>(l2_rW, pr, 256, CLSD);

    // ---- per-type input projection: 3 row-masked WMMA GEMMs, fused bias, h16 mirror ----
    k_gemm_wmma<<<dim3(gx, HIDD/16), gemmBlk, 0, stream>>>(xh, pfc0, h, h16, fc_b + 0*HIDD, ntype, 0, NN, IND, HIDD);
    k_gemm_wmma<<<dim3(gx, HIDD/16), gemmBlk, 0, stream>>>(xh, pfc1, h, h16, fc_b + 1*HIDD, ntype, 1, NN, IND, HIDD);
    k_gemm_wmma<<<dim3(gx, HIDD/16), gemmBlk, 0, stream>>>(xh, pfc2, h, h16, fc_b + 2*HIDD, ntype, 2, NN, IND, HIDD);

    // ================= layer 0: H=4, D=64, K=64 =================
    k_gemm_wmma<<<dim3(gx, 256/16), gemmBlk, 0, stream>>>(h16, p0, feat, nullptr, nullptr, nullptr, 0, NN, HIDD, 256);
    k_elr<<<gblk(NN*4, 256), 256, 0, stream>>>(feat, l0_al, l0_ar, elb, erb, NN, 4, 64);
    k_eetab<<<NETD, 4, 0, stream>>>(l0_Ee, l0_We, l0_ae, eet, 4, EDD);
    k_fill4<<<gblk(NN, 256), 256, 0, stream>>>((float4*)mbuf, -INFINITY, NN);
    k_fill4<<<gblk(NN, 256), 256, 0, stream>>>((float4*)sbuf, 0.f, NN);
    k_fill4<<<gblk((long long)NN*64, 256), 256, 0, stream>>>((float4*)agg, 0.f, (long long)NN*64);
    k_logit<<<gblk((long long)EE*4, 256), 256, 0, stream>>>(elb, erb, eet, src, dst, etype, elog, mbuf, EE, 4);
    k_expsum<<<gblk((long long)EE*4, 256), 256, 0, stream>>>(elog, mbuf, dst, sbuf, EE, 4);
    k_aggregate4<<<gblk((long long)EE*4*16, 256), 256, 0, stream>>>(elog, sbuf, src, dst, feat, agg,
                                                                    nullptr, attn0, EE, 4, 16);
    k_finalize<<<gblk((long long)NN*256, 256), 256, 0, stream>>>(agg, nullptr, l0_b, h, h16, NN, 256, 1);

    // ================= layer 1: H=4, D=64, K=256, residual=h0, res_attn=attn0 =================
    k_gemm_wmma<<<dim3(gx, 256/16), gemmBlk, 0, stream>>>(h16, p1, feat, nullptr, nullptr, nullptr, 0, NN, 256, 256);
    k_elr<<<gblk(NN*4, 256), 256, 0, stream>>>(feat, l1_al, l1_ar, elb, erb, NN, 4, 64);
    k_eetab<<<NETD, 4, 0, stream>>>(l1_Ee, l1_We, l1_ae, eet, 4, EDD);
    k_fill4<<<gblk(NN, 256), 256, 0, stream>>>((float4*)mbuf, -INFINITY, NN);
    k_fill4<<<gblk(NN, 256), 256, 0, stream>>>((float4*)sbuf, 0.f, NN);
    k_fill4<<<gblk((long long)NN*64, 256), 256, 0, stream>>>((float4*)agg, 0.f, (long long)NN*64);
    k_logit<<<gblk((long long)EE*4, 256), 256, 0, stream>>>(elb, erb, eet, src, dst, etype, elog, mbuf, EE, 4);
    k_expsum<<<gblk((long long)EE*4, 256), 256, 0, stream>>>(elog, mbuf, dst, sbuf, EE, 4);
    k_aggregate4<<<gblk((long long)EE*4*16, 256), 256, 0, stream>>>(elog, sbuf, src, dst, feat, agg,
                                                                    attn0, nullptr, EE, 4, 16);
    k_finalize<<<gblk((long long)NN*256, 256), 256, 0, stream>>>(agg, h, l1_b, h, h16, NN, 256, 1);

    // ================= layer 2: H=1, D=16, K=256, residual = h1 @ l2_resW, no act =================
    k_gemm_wmma<<<dim3(gx, CLSD/16), gemmBlk, 0, stream>>>(h16, p2, feat, nullptr, nullptr, nullptr, 0, NN, 256, CLSD);
    k_gemm_wmma<<<dim3(gx, CLSD/16), gemmBlk, 0, stream>>>(h16, pr, res2, nullptr, nullptr, nullptr, 0, NN, 256, CLSD);
    k_elr<<<gblk(NN, 256), 256, 0, stream>>>(feat, l2_al, l2_ar, elb, erb, NN, 1, CLSD);
    k_eetab<<<NETD, 1, 0, stream>>>(l2_Ee, l2_We, l2_ae, eet, 1, EDD);
    k_fill4<<<gblk(NN/4, 256), 256, 0, stream>>>((float4*)mbuf, -INFINITY, NN/4);
    k_fill4<<<gblk(NN/4, 256), 256, 0, stream>>>((float4*)sbuf, 0.f, NN/4);
    k_fill4<<<gblk((long long)NN*4, 256), 256, 0, stream>>>((float4*)agg, 0.f, (long long)NN*4);
    k_logit<<<gblk(EE, 256), 256, 0, stream>>>(elb, erb, eet, src, dst, etype, elog, mbuf, EE, 1);
    k_expsum<<<gblk(EE, 256), 256, 0, stream>>>(elog, mbuf, dst, sbuf, EE, 1);
    k_aggregate4<<<gblk((long long)EE*4, 256), 256, 0, stream>>>(elog, sbuf, src, dst, feat, agg,
                                                                 nullptr, nullptr, EE, 1, 4);
    k_finalize<<<gblk((long long)NN*CLSD, 256), 256, 0, stream>>>(agg, res2, l2_b, agg, nullptr, NN, CLSD, 0);

    // ---- row L2 normalize -> output ----
    k_norm<<<gblk(NN, 256), 256, 0, stream>>>(agg, (float*)d_out, NN, CLSD);
}
